// TopKSparseAutoencoder_27487790694626
// MI455X (gfx1250) — matrix operand order repaired and checked by reading, not compiled, not run on hardware
//
#include <hip/hip_runtime.h>
#include <hip/hip_bf16.h>

// ---------------- problem constants ----------------
#define D_IN   768
#define D_HID  16384
#define BATCH  8192
#define KMAX   32

// ---------------- encode GEMM tiling ---------------
#define KSTEP   32          // K per WMMA (bf16 16x16x32)
#define TILE_M  128         // rows of x per block
#define TILE_N  64          // hidden units per block
#define LDS_K   (KSTEP + 8) // pad to 40 halves (80B) -> conflict-free frag reads

// ---- gfx1250 async global->LDS path (guarded; falls back to sync copy) ----
#define AS1 __attribute__((address_space(1)))
#define AS3 __attribute__((address_space(3)))
#if defined(__AMDGCN__) && defined(__has_builtin)
#  if __has_builtin(__builtin_amdgcn_global_load_async_to_lds_b128) && \
      __has_builtin(__builtin_amdgcn_s_wait_asynccnt)
#    define HAVE_ASYNC_LDS 1
#  endif
#endif
#ifndef HAVE_ASYNC_LDS
#  define HAVE_ASYNC_LDS 0
#endif

typedef __attribute__((ext_vector_type(16))) __bf16 v16bf;
typedef __attribute__((ext_vector_type(8)))  float  v8f;
typedef __attribute__((ext_vector_type(4)))  int    v4i;

union FragU { v16bf v; uint4 q[2]; };

// copy 16 bytes global -> LDS (async DMA on gfx1250 if available)
__device__ __forceinline__ void cp16_g2l(const void* g, void* l) {
#if HAVE_ASYNC_LDS
    __builtin_amdgcn_global_load_async_to_lds_b128(
        (AS1 v4i*)g, (AS3 v4i*)l, 0, 0);
#else
    *reinterpret_cast<uint4*>(l) = *reinterpret_cast<const uint4*>(g);
#endif
}

__device__ __forceinline__ void wait_g2l() {
#if HAVE_ASYNC_LDS
    __builtin_amdgcn_s_wait_asynccnt(0);
#endif
}

// float -> bf16 (round-to-nearest-even), bit-level so we never do __bf16 scalar math
__device__ __forceinline__ unsigned short f2bf(float f) {
    unsigned int u = __float_as_uint(f);
    u += 0x7FFFu + ((u >> 16) & 1u);
    return (unsigned short)(u >> 16);
}

// order-preserving float -> uint mapping (for radix top-k select)
__device__ __forceinline__ unsigned int f2ord(float f) {
    unsigned int b = __float_as_uint(f);
    return (b & 0x80000000u) ? ~b : (b | 0x80000000u);
}

// ---------------------------------------------------------------------------
// Kernel 0a: xc_bf16 = bf16(x - b_dec)     [BATCH, D_IN]
// ---------------------------------------------------------------------------
__global__ __launch_bounds__(256) void prep_x_kernel(
    const float* __restrict__ x, const float* __restrict__ b_dec,
    unsigned short* __restrict__ xcb)
{
    size_t i = (size_t)blockIdx.x * blockDim.x + threadIdx.x;
    if (i >= (size_t)BATCH * D_IN) return;
    int col = (int)(i % D_IN);
    xcb[i] = f2bf(x[i] - b_dec[col]);
}

// ---------------------------------------------------------------------------
// Kernel 0b: W_enc -> bf16                  [D_HID, D_IN]
// ---------------------------------------------------------------------------
__global__ __launch_bounds__(256) void prep_w_kernel(
    const float* __restrict__ w, unsigned short* __restrict__ wb)
{
    size_t i = (size_t)blockIdx.x * blockDim.x + threadIdx.x;
    if (i >= (size_t)D_HID * D_IN) return;
    wb[i] = f2bf(w[i]);
}

__global__ void zero_loss_kernel(float* loss) { *loss = 0.0f; }

// ---------------------------------------------------------------------------
// Kernel 1: encode GEMM  z_pre = xc @ W_enc^T + b_enc  via WMMA bf16->f32
//   block = 256 threads = 8 waves; block tile 128(M) x 64(N)
//   wave grid 4(M) x 2(N); each wave: 32x32 = 2x2 tiles of 16x16, K-loop of 32
//   Global->LDS staging uses gfx1250 async-to-LDS DMA (ASYNCcnt), double buffered.
// ---------------------------------------------------------------------------
__global__ __launch_bounds__(256) void encode_wmma_kernel(
    const unsigned short* __restrict__ Xcb,   // [BATCH][D_IN] bf16 bits
    const unsigned short* __restrict__ Web,   // [D_HID][D_IN] bf16 bits
    const float* __restrict__ b_enc,
    float* __restrict__ Zpre)                 // [BATCH][D_HID]
{
    __shared__ __align__(16) unsigned short As[2][TILE_M][LDS_K];  // ~20 KB
    __shared__ __align__(16) unsigned short Bs[2][TILE_N][LDS_K];  // ~10 KB

    const int tid  = threadIdx.x;
    const int lane = tid & 31;
    const int wave = tid >> 5;
    const int m0 = blockIdx.y * TILE_M;
    const int n0 = blockIdx.x * TILE_N;
    const int wm = (wave & 3) << 5;   // wave M offset: 0/32/64/96
    const int wn = (wave >> 2) << 5;  // wave N offset: 0/32

    v8f acc[2][2] = {};

    // ---- cooperative global -> LDS tile fill (one K-step), async DMA ----
    auto load_tiles = [&](int buf, int ks) {
        // A: 128 rows x 32 halves; 2 threads/row, 16 halves (32B) each
        {
            int row  = tid >> 1;
            int half = tid & 1;
            const unsigned short* src = Xcb + (size_t)(m0 + row) * D_IN + ks + half * 16;
            unsigned short* dst = &As[buf][row][half * 16];
            cp16_g2l(src,     dst);
            cp16_g2l(src + 8, dst + 8);
            if (ks + 2 * KSTEP < D_IN)
                __builtin_prefetch(src + 2 * KSTEP, 0, 0);   // next-next A block
        }
        // B: 64 rows x 32 halves; 4 threads/row, 8 halves (16B) each
        {
            int row = tid >> 2;
            int q   = tid & 3;
            const unsigned short* src = Web + (size_t)(n0 + row) * D_IN + ks + q * 8;
            cp16_g2l(src, &Bs[buf][row][q * 8]);
            if (ks + 2 * KSTEP < D_IN)
                __builtin_prefetch(src + 2 * KSTEP, 0, 0);   // next-next B block
        }
    };

    // ---- fragment loads from LDS (contiguous k order k = 16*(lane/16) + s, same for A and B) ----
    // A 16x32: lanes 0-15 row M=lane hold K0..15
    //          lanes 16-31 same rows hold K16..31
    auto load_a = [&](int buf, int mt) -> v16bf {
        int row   = wm + (mt << 4) + (lane & 15);
        int khalf = lane >> 4;
        const unsigned short* p = &As[buf][row][0];
        FragU u;
        u.q[0] = *reinterpret_cast<const uint4*>(p + khalf * 16);
        u.q[1] = *reinterpret_cast<const uint4*>(p + khalf * 16 + 8);
        return u.v;
    };
    // B 32x16: lanes 0-15 col N=lane hold K0..15; lanes 16-31 hold K16..31
    auto load_b = [&](int buf, int nt) -> v16bf {
        int col   = wn + (nt << 4) + (lane & 15);
        int khalf = lane >> 4;
        const unsigned short* p = &Bs[buf][col][khalf * 16];
        FragU u;
        u.q[0] = *reinterpret_cast<const uint4*>(p);
        u.q[1] = *reinterpret_cast<const uint4*>(p + 8);
        return u.v;
    };

    load_tiles(0, 0);
    wait_g2l();
    __syncthreads();

    for (int ks = 0; ks < D_IN; ks += KSTEP) {
        const int buf = (ks / KSTEP) & 1;
        if (ks + KSTEP < D_IN) load_tiles(buf ^ 1, ks + KSTEP);  // async fill next buf

        v16bf a0 = load_a(buf, 0);
        v16bf a1 = load_a(buf, 1);
        v16bf b0 = load_b(buf, 0);
        v16bf b1 = load_b(buf, 1);

        acc[0][0] = __builtin_amdgcn_wmma_f32_16x16x32_bf16(
            false, a0, false, b0, (short)0, acc[0][0], false, false);
        acc[0][1] = __builtin_amdgcn_wmma_f32_16x16x32_bf16(
            false, a0, false, b1, (short)0, acc[0][1], false, false);
        acc[1][0] = __builtin_amdgcn_wmma_f32_16x16x32_bf16(
            false, a1, false, b0, (short)0, acc[1][0], false, false);
        acc[1][1] = __builtin_amdgcn_wmma_f32_16x16x32_bf16(
            false, a1, false, b1, (short)0, acc[1][1], false, false);

        wait_g2l();          // next buffer's DMA has landed in LDS
        __syncthreads();
    }

    // ---- epilogue: bias add + store (C/D layout: lane<16 -> M=r, lane>=16 -> M=r+8)
    #pragma unroll
    for (int mt = 0; mt < 2; ++mt) {
        #pragma unroll
        for (int nt = 0; nt < 2; ++nt) {
            int n     = n0 + wn + (nt << 4) + (lane & 15);
            float bia = b_enc[n];
            int mbase = m0 + wm + (mt << 4) + ((lane >> 4) << 3);
            v8f c = acc[mt][nt];
            #pragma unroll
            for (int r = 0; r < 8; ++r)
                Zpre[(size_t)(mbase + r) * D_HID + n] = c[r] + bia;
        }
    }
}

// ---------------------------------------------------------------------------
// Kernel 2: per-row exact top-k (radix select on order-mapped uints),
//           in-place scatter relu(top-k) into z, record (idx,val) lists.
//           One 256-thread block per row; each thread keeps 64 values in regs.
// ---------------------------------------------------------------------------
__global__ __launch_bounds__(256) void topk_kernel(
    float* __restrict__ Z,            // [BATCH][D_HID], z_pre in / sparse z out
    int*   __restrict__ idx_out,      // [BATCH][KMAX]
    float* __restrict__ val_out,      // [BATCH][KMAX]
    const int* __restrict__ kp)
{
    __shared__ unsigned int red[256];
    __shared__ unsigned int scount;

    const int b   = blockIdx.x;
    const int tid = threadIdx.x;
    const int k   = min(*kp, KMAX);
    float* zrow = Z + (size_t)b * D_HID;

    unsigned int myu[64];     // D_HID = 64 * 256
    #pragma unroll
    for (int j = 0; j < 64; ++j)
        myu[j] = f2ord(zrow[tid + (j << 8)]);

    // bitwise binary search for the k-th largest mapped value
    unsigned int prefix = 0u;
    for (int bit = 31; bit >= 0; --bit) {
        unsigned int test = prefix | (1u << bit);
        unsigned int c = 0;
        #pragma unroll
        for (int j = 0; j < 64; ++j) c += (myu[j] >= test) ? 1u : 0u;
        red[tid] = c;
        __syncthreads();
        for (int s = 128; s > 0; s >>= 1) {
            if (tid < s) red[tid] += red[tid + s];
            __syncthreads();
        }
        if (red[0] >= (unsigned int)k) prefix = test;
        __syncthreads();
    }

    if (tid == 0) scount = 0u;
    __syncthreads();

    // scatter: keep first k elements >= threshold, relu them, zero the rest
    #pragma unroll
    for (int j = 0; j < 64; ++j) {
        int i = tid + (j << 8);
        float outv = 0.0f;
        if (myu[j] >= prefix) {
            unsigned int slot = atomicAdd(&scount, 1u);
            if (slot < (unsigned int)k) {
                float v = zrow[i];
                outv = v > 0.0f ? v : 0.0f;
                idx_out[b * KMAX + slot] = i;
                val_out[b * KMAX + slot] = outv;
            }
        }
        zrow[i] = outv;
    }
}

// ---------------------------------------------------------------------------
// Kernel 3: sparse decode x_hat = z @ W_dec^T + b_dec (32 nnz/row) + MSE loss
// ---------------------------------------------------------------------------
__global__ __launch_bounds__(256) void decode_kernel(
    const float* __restrict__ x,
    const int*   __restrict__ idx_in,
    const float* __restrict__ val_in,
    const float* __restrict__ W_dec,   // [D_IN][D_HID]
    const float* __restrict__ b_dec,
    float* __restrict__ x_hat,
    float* __restrict__ loss,
    const int* __restrict__ kp)
{
    __shared__ int   sidx[KMAX];
    __shared__ float sval[KMAX];
    __shared__ float sred[256];

    const int b   = blockIdx.x;
    const int tid = threadIdx.x;
    const int k   = min(*kp, KMAX);

    if (tid < KMAX) {
        sidx[tid] = idx_in[b * KMAX + tid];
        sval[tid] = val_in[b * KMAX + tid];
    }
    __syncthreads();

    float lsum = 0.0f;
    for (int i = tid; i < D_IN; i += 256) {
        float accv = b_dec[i];
        const float* wrow = W_dec + (size_t)i * D_HID;
        for (int j = 0; j < k; ++j)
            accv = fmaf(sval[j], wrow[sidx[j]], accv);
        x_hat[(size_t)b * D_IN + i] = accv;
        float d = accv - x[(size_t)b * D_IN + i];
        lsum = fmaf(d, d, lsum);
    }

    sred[tid] = lsum;
    __syncthreads();
    for (int s = 128; s > 0; s >>= 1) {
        if (tid < s) sred[tid] += sred[tid + s];
        __syncthreads();
    }
    if (tid == 0)
        atomicAdd(loss, sred[0] * (1.0f / ((float)BATCH * (float)D_IN)));
}

// ---------------------------------------------------------------------------
extern "C" void kernel_launch(void* const* d_in, const int* in_sizes, int n_in,
                              void* d_out, int out_size, void* d_ws, size_t ws_size,
                              hipStream_t stream) {
    const float* x     = (const float*)d_in[0];
    const float* W_enc = (const float*)d_in[1];
    const float* b_enc = (const float*)d_in[2];
    const float* W_dec = (const float*)d_in[3];
    const float* b_dec = (const float*)d_in[4];
    const int*   kp    = (const int*)d_in[5];

    // d_out = [x_hat (B*D_IN)] [z (B*D_HID)] [loss (1)]
    float* xhat = (float*)d_out;
    float* z    = xhat + (size_t)BATCH * D_IN;
    float* loss = z + (size_t)BATCH * D_HID;

    // workspace layout (~40 MB)
    char* ws = (char*)d_ws;
    size_t off = 0;
    unsigned short* Xcb = (unsigned short*)(ws + off); off += (size_t)BATCH * D_IN * 2;   // 12.6 MB
    unsigned short* Web = (unsigned short*)(ws + off); off += (size_t)D_HID * D_IN * 2;   // 25.2 MB
    int*   idxs = (int*)(ws + off);                    off += (size_t)BATCH * KMAX * 4;   // 1 MB
    float* vals = (float*)(ws + off);                  off += (size_t)BATCH * KMAX * 4;   // 1 MB
    (void)ws_size; (void)in_sizes; (void)n_in; (void)out_size;

    // 0) prep
    {
        int nx = BATCH * D_IN;
        prep_x_kernel<<<(nx + 255) / 256, 256, 0, stream>>>(x, b_dec, Xcb);
        int nw = D_HID * D_IN;
        prep_w_kernel<<<(nw + 255) / 256, 256, 0, stream>>>(W_enc, Web);
        zero_loss_kernel<<<1, 1, 0, stream>>>(loss);
    }

    // 1) encode GEMM (WMMA bf16): z_pre staged into the z region of d_out
    {
        dim3 grid(D_HID / TILE_N, BATCH / TILE_M);   // 256 x 64 blocks
        encode_wmma_kernel<<<grid, 256, 0, stream>>>(Xcb, Web, b_enc, z);
    }

    // 2) exact per-row top-k + relu scatter (rewrites z in place)
    topk_kernel<<<BATCH, 256, 0, stream>>>(z, idxs, vals, kp);

    // 3) sparse decode + loss
    decode_kernel<<<BATCH, 256, 0, stream>>>(x, idxs, vals, W_dec, b_dec,
                                             xhat, loss, kp);
}
